// mLSTMCell_21345987461771
// MI455X (gfx1250) — compile-verified, matrix operand order checked by reading, not run on hardware
//
#include <hip/hip_runtime.h>
#include <hip/hip_bf16.h>
#include <math.h>

// Problem sizes (compile-time constants from the reference)
#define B_SZ  64
#define IN_SZ 512
#define M_SZ  1024

typedef __attribute__((ext_vector_type(16))) _Float16 v16h;
typedef __attribute__((ext_vector_type(8)))  _Float16 v8h;
typedef __attribute__((ext_vector_type(8)))  float    v8f;
typedef __attribute__((ext_vector_type(8)))  float    f8;
typedef __attribute__((ext_vector_type(4)))  float    f4;

// Load 16 f32 (two aligned 8-float runs: [p..p+7] and [p+16..p+23]) and pack
// as the per-lane f16 WMMA fragment (K pattern per CDNA5 ISA:
// lanes 0-15 hold K={0..7,16..23}, lanes 16-31 hold K={8..15,24..31} -- the
// caller offsets p by half*8 so both halves use this same helper).
__device__ __forceinline__ v16h load_frag_f16(const float* __restrict__ p) {
    f8 lo = *(const f8*)(p);        // 32B aligned: rows are 512 f32, offsets %8==0
    f8 hi = *(const f8*)(p + 16);
    v8h hlo = __builtin_convertvector(lo, v8h);
    v8h hhi = __builtin_convertvector(hi, v8h);
    return __builtin_shufflevector(hlo, hhi,
                                   0, 1, 2, 3, 4, 5, 6, 7,
                                   8, 9, 10, 11, 12, 13, 14, 15);
}

// ---------------------------------------------------------------------------
// Kernel 1: 6 gate projections  pre[g][b][m] = sum_k x[b][k]*W_g[m][k] + bias
// One wave = one 16(b) x 16(m) tile, K swept 32 at a time with
// v_wmma_f32_16x16x32_f16 (x4 unrolled). Activation chosen by a single
// wave-uniform scalar branch on g (blockIdx.z) so pointers stay in SGPRs.
// grid = (M/64, B/16, 6), block = 128 (4 waves, each a different m-tile).
// ---------------------------------------------------------------------------
__global__ void __launch_bounds__(128)
gates_gemm_wmma(const float* __restrict__ x,
                const float* __restrict__ Wq, const float* __restrict__ Bq,
                const float* __restrict__ Wk, const float* __restrict__ Bk,
                const float* __restrict__ Wv, const float* __restrict__ Bv,
                const float* __restrict__ Wi, const float* __restrict__ Bi,
                const float* __restrict__ Wf, const float* __restrict__ Bf,
                const float* __restrict__ Wo, const float* __restrict__ Bo,
                float* __restrict__ gates)
{
    const int g    = blockIdx.z;        // wave-uniform gate index
    const int wid  = threadIdx.x >> 5;
    const int lane = threadIdx.x & 31;
    const int hv   = lane >> 4;         // which 16-lane half
    const int row  = lane & 15;

    const int b0 = blockIdx.y << 4;
    const int m0 = ((blockIdx.x << 2) + wid) << 4;

    // Uniform scalar selection: keeps W/bias base addresses in SGPRs
    // (no movrel / flat_load pointer gymnastics).
    const float* __restrict__ W;
    const float* __restrict__ Bp;
    switch (g) {
        case 0:  W = Wq; Bp = Bq; break;
        case 1:  W = Wk; Bp = Bk; break;
        case 2:  W = Wv; Bp = Bv; break;
        case 3:  W = Wi; Bp = Bi; break;
        case 4:  W = Wf; Bp = Bf; break;
        default: W = Wo; Bp = Bo; break;
    }

    const float* __restrict__ xrow = x + (size_t)(b0 + row) * IN_SZ;  // A: M=b rows
    const float* __restrict__ wrow = W + (size_t)(m0 + row) * IN_SZ;  // B: N=m cols (K-major)
    const int kh = hv << 3;

    v8f acc = {};
#pragma unroll 4
    for (int k0 = 0; k0 < IN_SZ; k0 += 32) {
        const int ka = k0 + kh;
        v16h afrag = load_frag_f16(xrow + ka);
        v16h bfrag = load_frag_f16(wrow + ka);
        acc = __builtin_amdgcn_wmma_f32_16x16x32_f16(
            /*neg_a=*/false, afrag, /*neg_b=*/false, bfrag,
            /*c_mod=*/(short)0, acc, /*reuse_a=*/false, /*reuse_b=*/false);
    }

    // D layout: VGPR r -> out_b = b0 + hv*8 + r, out_m = m0 + (lane&15)
    const int   colm = m0 + row;
    const float bias = Bp[colm];
    float* __restrict__ dst =
        gates + (size_t)g * B_SZ * M_SZ + (size_t)(b0 + (hv << 3)) * M_SZ + colm;

    float pre[8];
#pragma unroll
    for (int r = 0; r < 8; ++r) pre[r] = acc[r] + bias;

    // Uniform branch on g hoisted OUT of the store loop: four clean epilogues.
    if (g == 3) {                       // i: exponential input gate
#pragma unroll
        for (int r = 0; r < 8; ++r) dst[(size_t)r * M_SZ] = __expf(pre[r]);
    } else if (g >= 4) {                // f, o: sigmoid
#pragma unroll
        for (int r = 0; r < 8; ++r)
            dst[(size_t)r * M_SZ] = 1.0f / (1.0f + __expf(-pre[r]));
    } else if (g == 1) {                // k: scale by 1/sqrt(1024)
#pragma unroll
        for (int r = 0; r < 8; ++r) dst[(size_t)r * M_SZ] = pre[r] * 0.03125f;
    } else {                            // q, v: identity
#pragma unroll
        for (int r = 0; r < 8; ++r) dst[(size_t)r * M_SZ] = pre[r];
    }
}

// ---------------------------------------------------------------------------
// Kernel 2: n_t = f*n_prev + i*k  and  denom[b] = max(n_t . q, 1)
// One 256-thread block per b; float4 per thread; LDS tree reduction.
// ---------------------------------------------------------------------------
__global__ void __launch_bounds__(256)
nt_denom_kernel(const float* __restrict__ gates,
                const float* __restrict__ n_prev,
                float* __restrict__ nt_out,
                float* __restrict__ denom)
{
    const int b = blockIdx.x;
    const int t = threadIdx.x;

    const float* qv = gates + (size_t)(0 * B_SZ + b) * M_SZ;
    const float* kv = gates + (size_t)(1 * B_SZ + b) * M_SZ;
    const float* iv = gates + (size_t)(3 * B_SZ + b) * M_SZ;
    const float* fv = gates + (size_t)(4 * B_SZ + b) * M_SZ;
    const float* np = n_prev + (size_t)b * M_SZ;

    f4 f4v = *((const f4*)fv + t);
    f4 i4  = *((const f4*)iv + t);
    f4 k4  = *((const f4*)kv + t);
    f4 q4  = *((const f4*)qv + t);
    f4 n4  = *((const f4*)np + t);

    f4 nt = f4v * n4 + i4 * k4;
    *((f4*)(nt_out + (size_t)b * M_SZ) + t) = nt;

    f4 pr = nt * q4;
    float a = pr[0] + pr[1] + pr[2] + pr[3];

    __shared__ float red[256];
    red[t] = a;
    __syncthreads();
    for (int s = 128; s > 0; s >>= 1) {
        if (t < s) red[t] += red[t + s];
        __syncthreads();
    }
    if (t == 0) denom[b] = fmaxf(red[0], 1.0f);
}

// ---------------------------------------------------------------------------
// Kernel 3 (bandwidth-bound, ~22us roofline on 23.3 TB/s): per (b,m) row:
//   C_t[n] = f[m]*C_prev[n] + (i[m]*v[m])*k[n]      (streamed, NT hints)
//   h_t[m] = o[m] * (sum_n C_t[n]*q[n]) / denom[b]  (fused row-dot)
// grid = (M, B), block = 256; one float4 of the row per thread.
// ---------------------------------------------------------------------------
__global__ void __launch_bounds__(256)
c_update_kernel(const float* __restrict__ gates,
                const float* __restrict__ C_prev,
                const float* __restrict__ denom,
                float* __restrict__ C_out,
                float* __restrict__ h_out)
{
    const int m = blockIdx.x;
    const int b = blockIdx.y;
    const int t = threadIdx.x;

    const float* qv = gates + (size_t)(0 * B_SZ + b) * M_SZ;
    const float* kv = gates + (size_t)(1 * B_SZ + b) * M_SZ;
    const float vm  = gates[(size_t)(2 * B_SZ + b) * M_SZ + m];
    const float im  = gates[(size_t)(3 * B_SZ + b) * M_SZ + m];
    const float fm  = gates[(size_t)(4 * B_SZ + b) * M_SZ + m];
    const float om  = gates[(size_t)(5 * B_SZ + b) * M_SZ + m];
    const float ivm = im * vm;

    const size_t base = ((size_t)b * M_SZ + m) * M_SZ;

    // Single-use 512MB stream: non-temporal so it doesn't thrash L2.
    f4 cp = __builtin_nontemporal_load((const f4*)(C_prev + base) + t);
    f4 k4 = *((const f4*)kv + t);   // 4KB row, L2/L0 resident across m-blocks
    f4 q4 = *((const f4*)qv + t);

    f4 ct = fm * cp + ivm * k4;
    __builtin_nontemporal_store(ct, (f4*)(C_out + base) + t);

    f4 pr = ct * q4;
    float a = pr[0] + pr[1] + pr[2] + pr[3];

    __shared__ float red[256];
    red[t] = a;
    __syncthreads();
    for (int s = 128; s > 0; s >>= 1) {
        if (t < s) red[t] += red[t + s];
        __syncthreads();
    }
    if (t == 0) h_out[(size_t)b * M_SZ + m] = om * red[0] / denom[b];
}

// ---------------------------------------------------------------------------
// d_in order: x, h_prev, c_prev, C_prev, n_prev, m_prev,
//             Wq,bq, Wk,bk, Wv,bv, Wi,bi, Wf,bf, Wo,bo
// d_out: h_t [64*1024] | C_t [64*1024*1024] | n_t [64*1024]
// ---------------------------------------------------------------------------
extern "C" void kernel_launch(void* const* d_in, const int* in_sizes, int n_in,
                              void* d_out, int out_size, void* d_ws, size_t ws_size,
                              hipStream_t stream) {
    (void)in_sizes; (void)n_in; (void)out_size; (void)ws_size;

    const float* x      = (const float*)d_in[0];
    const float* C_prev = (const float*)d_in[3];
    const float* n_prev = (const float*)d_in[4];
    const float* Wq = (const float*)d_in[6];  const float* bq = (const float*)d_in[7];
    const float* Wk = (const float*)d_in[8];  const float* bk = (const float*)d_in[9];
    const float* Wv = (const float*)d_in[10]; const float* bv = (const float*)d_in[11];
    const float* Wi = (const float*)d_in[12]; const float* bi = (const float*)d_in[13];
    const float* Wf = (const float*)d_in[14]; const float* bf = (const float*)d_in[15];
    const float* Wo = (const float*)d_in[16]; const float* bo = (const float*)d_in[17];

    float* gates = (float*)d_ws;                         // [6][64][1024] f32 = 1.5MB
    float* denom = gates + (size_t)6 * B_SZ * M_SZ;      // [64] f32

    float* h_out  = (float*)d_out;
    float* C_out  = h_out + (size_t)B_SZ * M_SZ;
    float* nt_out = C_out + (size_t)B_SZ * M_SZ * M_SZ;

    gates_gemm_wmma<<<dim3(M_SZ / 64, B_SZ / 16, 6), 128, 0, stream>>>(
        x, Wq, bq, Wk, bk, Wv, bv, Wi, bi, Wf, bf, Wo, bo, gates);

    nt_denom_kernel<<<dim3(B_SZ), 256, 0, stream>>>(gates, n_prev, nt_out, denom);

    c_update_kernel<<<dim3(M_SZ, B_SZ), 256, 0, stream>>>(
        gates, C_prev, denom, C_out, h_out);
}
